// HardAttn_53094385713488
// MI455X (gfx1250) — matrix operand order repaired and checked by reading, not compiled, not run on hardware
//
#include <hip/hip_runtime.h>

// ---------------------------------------------------------------------------
// HardAttn for MI455X (gfx1250, wave32, WMMA).
// logits GEMM on v_wmma_f32_16x16x32_bf16 with a 3-term bf16 hi/lo split
// (qhi*khi + qhi*klo + qlo*khi) for ~2^-16 effective precision, fused
// online-softmax-entropy + argmax + v-gather. Logits never touch memory.
// k is double-buffered into LDS with async global->LDS copies (ASYNCcnt).
// ---------------------------------------------------------------------------

#define NH   16
#define NM   512
#define ND   64
#define NL   2048
#define NB   4
#define REPS 1.1920929e-07f

#define OUT_ELEMS   (NB * NH * ND * NL)        // 8388608
#define ENT_POS     OUT_ELEMS                  // 8388608
#define IDX_BASE    (OUT_ELEMS + 1)            // 8388609
#define NROWS_TOT   (NB * NH * NL)             // 131072

typedef __attribute__((ext_vector_type(16))) __bf16 bf16x16;
typedef __attribute__((ext_vector_type(8)))  __bf16 bf16x8;
typedef __attribute__((ext_vector_type(2)))  __bf16 bf16x2;
typedef __attribute__((ext_vector_type(8)))  float  f32x8;
typedef int v4i_gv __attribute__((vector_size(16)));   // matches builtin param

// workspace layout (bytes)
#define WS_KHI 0u
#define WS_KLO 1048576u
#define WS_VN  2097152u
#define WS_ENT 4194304u   // 1024 floats of per-block entropy partials

#if __has_builtin(__builtin_amdgcn_global_load_async_to_lds_b128)
#define HAS_ASYNC_LDS 1
#else
#define HAS_ASYNC_LDS 0
#endif

// ---------------------------------------------------------------------------
// Prep: rms-norm kv. k -> bf16 hi/lo pair, v -> f32. One wave32 per row of 64.
// ---------------------------------------------------------------------------
__global__ __launch_bounds__(256) void hardattn_prep(
    const float* __restrict__ kv,
    __bf16* __restrict__ khi, __bf16* __restrict__ klo,
    float* __restrict__ vn) {
  const int row  = blockIdx.x * 8 + (threadIdx.x >> 5);   // 0..16383
  const int lane = threadIdx.x & 31;
  const float2 vv = *(const float2*)(kv + (size_t)row * ND + lane * 2);
  float ss = vv.x * vv.x + vv.y * vv.y;
  #pragma unroll
  for (int m = 1; m < 32; m <<= 1) ss += __shfl_xor(ss, m);
  const float r = rsqrtf(ss * (1.0f / ND) + REPS);
  const float a = vv.x * r, b = vv.y * r;
  if (row < NH * NM) {                       // k rows
    const __bf16 ah = (__bf16)a, bh = (__bf16)b;
    bf16x2 hi; hi[0] = ah; hi[1] = bh;
    bf16x2 lo; lo[0] = (__bf16)(a - (float)ah); lo[1] = (__bf16)(b - (float)bh);
    *(bf16x2*)(khi + (size_t)row * ND + lane * 2) = hi;
    *(bf16x2*)(klo + (size_t)row * ND + lane * 2) = lo;
  } else {                                   // v rows
    float2 o; o.x = a; o.y = b;
    *(float2*)(vn + (size_t)(row - NH * NM) * ND + lane * 2) = o;
  }
}

// ---------------------------------------------------------------------------
// Fragment / copy helpers
// ---------------------------------------------------------------------------
__device__ inline bf16x16 ld_frag(const __bf16* p0, const __bf16* p1) {
  bf16x8 a = *(const bf16x8*)p0;
  bf16x8 b = *(const bf16x8*)p1;
  bf16x16 r;
  #pragma unroll
  for (int i = 0; i < 8; ++i) { r[i] = a[i]; r[i + 8] = b[i]; }
  return r;
}

__device__ inline f32x8 wmma_bf16(bf16x16 a, bf16x16 b, f32x8 c) {
  return __builtin_amdgcn_wmma_f32_16x16x32_bf16(
      false, a, false, b, (short)0, c, false, false);
}

// 16B global -> LDS copy; async (ASYNCcnt-tracked) when the builtin exists.
__device__ inline void cp16_g2l(const __bf16* __restrict__ g, __bf16* l) {
#if HAS_ASYNC_LDS
  __builtin_amdgcn_global_load_async_to_lds_b128(
      (__attribute__((address_space(1))) v4i_gv*)g,
      (__attribute__((address_space(3))) v4i_gv*)l, 0, 0);
#else
  *(uint4*)l = *(const uint4*)g;
#endif
}

__device__ inline void async_wait0() {
#if HAS_ASYNC_LDS
  asm volatile("s_wait_asynccnt 0" ::: "memory");
#endif
}

// ---------------------------------------------------------------------------
// Main fused kernel. grid = (16 l-tiles, 64 bh). block = 256 (8 waves).
// ---------------------------------------------------------------------------
__global__ __launch_bounds__(256) void hardattn_main(
    const float* __restrict__ x,
    const __bf16* __restrict__ khi_g, const __bf16* __restrict__ klo_g,
    const float* __restrict__ vn,
    float* __restrict__ out, float* __restrict__ ent_partial) {
  // LDS map (bytes). Region0 is reused: qraw (phase A/B) then two k buffers:
  //   buf b @ b*36864: khi [128][72] (18432 B) then klo [128][72].
  __shared__ __align__(16) char smem[112672];
  float*  qraw  = (float*)smem;                 // [128][65] f32 (phase A/B)
  float*  sums  = (float*)(smem + 73728);       // [128][2]
  float*  rsq   = (float*)(smem + 74752);       // [128]
  __bf16* qh    = (__bf16*)(smem + 75264);      // [128][72]
  __bf16* ql    = (__bf16*)(smem + 93696);      // [128][72]
  int*    idx_s = (int*)(smem + 112128);        // [128]
  float*  ent_s = (float*)(smem + 112640);      // [8]

  const int tid = threadIdx.x;
  const int bh  = blockIdx.y;            // b*NH + h
  const int h   = bh & (NH - 1);
  const int l0  = blockIdx.x * 128;

  // ---- Phase A: stream x tile (coalesced over l), transpose into LDS ----
  {
    const int l = tid & 127, half = tid >> 7;
    float ss = 0.f;
    #pragma unroll 8
    for (int i = 0; i < 32; ++i) {
      const int d = half * 32 + i;
      const float v = x[(size_t)(bh * ND + d) * NL + l0 + l];
      qraw[l * 65 + d] = v;
      ss += v * v;
    }
    sums[l * 2 + half] = ss;
  }
  __syncthreads();
  if (tid < 128)
    rsq[tid] = rsqrtf((sums[tid * 2] + sums[tid * 2 + 1]) * (1.f / ND) + REPS);
  __syncthreads();

  // ---- Phase B: rms-norm -> bf16 hi/lo A-tiles (144B padded rows) ----
  for (int j = tid; j < 128 * ND; j += 256) {
    const int l = j >> 6, d = j & 63;
    const float v = qraw[l * 65 + d] * rsq[l];
    const __bf16 hi = (__bf16)v;
    qh[l * 72 + d] = hi;
    ql[l * 72 + d] = (__bf16)(v - (float)hi);
  }
  __syncthreads();   // qh/ql ready; qraw region free for k buffers

  const int wave = tid >> 5, lane = tid & 31;
  const int hsel = lane >> 4, nsel = lane & 15;

  // ---- A fragments (per wave: 16 queries), documented 16-bit A layout ----
  bf16x16 aHi[2], aLo[2];
  {
    const int lrow = wave * 16 + nsel;
    #pragma unroll
    for (int kh = 0; kh < 2; ++kh) {
      const int d0 = kh * 32 + hsel * 8;
      aHi[kh] = ld_frag(qh + lrow * 72 + d0, qh + lrow * 72 + d0 + 16);
      aLo[kh] = ld_frag(ql + lrow * 72 + d0, ql + lrow * 72 + d0 + 16);
    }
  }

  float mx[8], sE[8], tE[8], bv[8];
  int bi[8];
  #pragma unroll
  for (int r = 0; r < 8; ++r) {
    mx[r] = -__builtin_inff(); sE[r] = 0.f; tE[r] = 0.f;
    bv[r] = -__builtin_inff(); bi[r] = 0;
  }

  // k chunk staging: 128 codes (hi+lo) into buffer `buf` (2048 x 16B units)
  auto stage = [&](int chunk, int buf) {
    char* base = smem + buf * 36864;
    for (int u = tid; u < 2048; u += 256) {
      const int arr = u >> 10, j = u & 1023;
      const int code = j >> 3, du = j & 7;
      const __bf16* src = (arr ? klo_g : khi_g) +
          ((size_t)(h * NM + chunk * 128 + code) * ND + du * 8);
      __bf16* dst = (__bf16*)(base + arr * 18432 + code * 144 + du * 16);
      cp16_g2l(src, dst);
    }
  };

  // ---- Main loop: 4 chunks x 128 codes, double-buffered async staging ----
  stage(0, 0);
  async_wait0();
  __syncthreads();
  for (int chunk = 0; chunk < 4; ++chunk) {
    const int buf = chunk & 1;
    if (chunk + 1 < 4) stage(chunk + 1, buf ^ 1);   // prefetch next chunk
    const __bf16* khi_s = (const __bf16*)(smem + buf * 36864);
    const __bf16* klo_s = khi_s + 18432 / 2;

    // two code-tiles per iteration -> two independent WMMA chains
    for (int ntl = 0; ntl < 8; ntl += 2) {
      f32x8 c0 = {0.f, 0.f, 0.f, 0.f, 0.f, 0.f, 0.f, 0.f};
      f32x8 c1 = {0.f, 0.f, 0.f, 0.f, 0.f, 0.f, 0.f, 0.f};
      #pragma unroll
      for (int kh = 0; kh < 2; ++kh) {
        const int d0 = kh * 32 + hsel * 8;           // 16-bit B layout
        const __bf16* kh_p0 = khi_s + ((ntl + 0) * 16 + nsel) * 72 + d0;
        const __bf16* kl_p0 = klo_s + ((ntl + 0) * 16 + nsel) * 72 + d0;
        const __bf16* kh_p1 = khi_s + ((ntl + 1) * 16 + nsel) * 72 + d0;
        const __bf16* kl_p1 = klo_s + ((ntl + 1) * 16 + nsel) * 72 + d0;
        const bf16x16 bHi0 = ld_frag(kh_p0, kh_p0 + 16);
        const bf16x16 bLo0 = ld_frag(kl_p0, kl_p0 + 16);
        const bf16x16 bHi1 = ld_frag(kh_p1, kh_p1 + 16);
        const bf16x16 bLo1 = ld_frag(kl_p1, kl_p1 + 16);
        c0 = wmma_bf16(aHi[kh], bHi0, c0);
        c1 = wmma_bf16(aHi[kh], bHi1, c1);
        c0 = wmma_bf16(aHi[kh], bLo0, c0);
        c1 = wmma_bf16(aHi[kh], bLo1, c1);
        c0 = wmma_bf16(aLo[kh], bHi0, c0);
        c1 = wmma_bf16(aLo[kh], bHi1, c1);
      }
      #pragma unroll
      for (int half = 0; half < 2; ++half) {
        const f32x8 c = half ? c1 : c0;
        const int code = (chunk * 8 + ntl + half) * 16 + nsel;
        #pragma unroll
        for (int r = 0; r < 8; ++r) {
          const float v = c[r];
          if (v > bv[r]) { bv[r] = v; bi[r] = code; }
          const float m2 = fmaxf(mx[r], v);
          const float e1 = __expf(mx[r] - m2);
          const float e2 = __expf(v - m2);
          sE[r] = sE[r] * e1 + e2;
          tE[r] = tE[r] * e1 + v * e2;
          mx[r] = m2;
        }
      }
    }
    async_wait0();     // next chunk landed (this wave's portion)
    __syncthreads();   // all waves: reads done + all portions visible
  }

  // ---- Butterfly merge across the 16 column-lanes ----
  #pragma unroll
  for (int m = 1; m <= 8; m <<= 1) {
    #pragma unroll
    for (int r = 0; r < 8; ++r) {
      const float omx = __shfl_xor(mx[r], m);
      const float os  = __shfl_xor(sE[r], m);
      const float ot  = __shfl_xor(tE[r], m);
      const float obv = __shfl_xor(bv[r], m);
      const int   obi = __shfl_xor(bi[r], m);
      const float m2 = fmaxf(mx[r], omx);
      const float e1 = __expf(mx[r] - m2);
      const float e2 = __expf(omx - m2);
      sE[r] = sE[r] * e1 + os * e2;
      tE[r] = tE[r] * e1 + ot * e2;
      mx[r] = m2;
      if (obv > bv[r] || (obv == bv[r] && obi < bi[r])) { bv[r] = obv; bi[r] = obi; }
    }
  }

  // ---- Entropy partial + indices ----
  float ent = 0.f;
  #pragma unroll
  for (int r = 0; r < 8; ++r)
    ent += mx[r] + __logf(sE[r]) - tE[r] / sE[r];   // LSE - E[logit]
  ent += __shfl_xor(ent, 16);
  if (lane == 0) ent_s[wave] = ent;
  if (nsel == 0) {
    #pragma unroll
    for (int r = 0; r < 8; ++r) {
      const int l = wave * 16 + hsel * 8 + r;
      idx_s[l] = bi[r];
      out[(size_t)IDX_BASE + (size_t)bh * NL + l0 + l] = (float)bi[r];
    }
  }
  __syncthreads();
  if (tid == 0) {
    float t = 0.f;
    #pragma unroll
    for (int w = 0; w < 8; ++w) t += ent_s[w];
    ent_partial[blockIdx.y * 16 + blockIdx.x] = t * (1.f / NROWS_TOT);
  }

  // ---- Gather v[argmax], coalesced over l (v slice is L2-resident) ----
  for (int i = tid; i < 8192; i += 256) {
    const int d = i >> 7, l = i & 127;
    const int code = idx_s[l];
    out[(size_t)(bh * ND + d) * NL + l0 + l] =
        vn[((size_t)h * NM + code) * ND + d];
  }
}

// ---------------------------------------------------------------------------
// Deterministic entropy reduction (1024 partials -> scalar)
// ---------------------------------------------------------------------------
__global__ __launch_bounds__(256) void hardattn_entred(
    const float* __restrict__ part, float* __restrict__ out) {
  __shared__ float red[256];
  float s = 0.f;
  for (int i = threadIdx.x; i < 1024; i += 256) s += part[i];
  red[threadIdx.x] = s;
  __syncthreads();
  for (int st = 128; st > 0; st >>= 1) {
    if (threadIdx.x < st) red[threadIdx.x] += red[threadIdx.x + st];
    __syncthreads();
  }
  if (threadIdx.x == 0) out[ENT_POS] = red[0];
}

extern "C" void kernel_launch(void* const* d_in, const int* in_sizes, int n_in,
                              void* d_out, int out_size, void* d_ws, size_t ws_size,
                              hipStream_t stream) {
  const float* x  = (const float*)d_in[0];   // [B, H*D, L] f32
  const float* kv = (const float*)d_in[1];   // [2, H, M, D] f32
  float* out = (float*)d_out;
  char*  ws  = (char*)d_ws;
  __bf16* khi = (__bf16*)(ws + WS_KHI);
  __bf16* klo = (__bf16*)(ws + WS_KLO);
  float*  vn  = (float*)(ws + WS_VN);
  float*  ent = (float*)(ws + WS_ENT);

  hardattn_prep<<<2048, 256, 0, stream>>>(kv, khi, klo, vn);
  dim3 grid(NL / 128, NB * NH);
  hardattn_main<<<grid, 256, 0, stream>>>(x, khi, klo, vn, out, ent);
  hardattn_entred<<<1, 256, 0, stream>>>(ent, out);
}
